// CWVAE_14224931684863
// MI455X (gfx1250) — matrix operand order repaired and checked by reading, not compiled, not run on hardware
//
#include <hip/hip_runtime.h>
#include <cstddef>

// ---------------------------------------------------------------------------
// CWVAE hierarchical RSSM unroll for gfx1250 (MI455X).
//   * Dead-code: prior branch (pm_w/ps_w/noise_p) does not affect d_out -> skipped.
//   * One persistent kernel per level; grid = 4 blocks, each block owns 16
//     batch rows (one WMMA M-tile) and carries the recurrence privately.
//   * All GEMMs via v_wmma_f32_16x16x32_f16 (f16 inputs, f32 accumulate).
//   * Weights converted f32 -> f16 *transposed* (N x K) into workspace so
//     B-fragments are contiguous 32B/lane loads (L2-resident, 192MB L2).
//   * Activations staged in LDS (dynamic, ~68KB/WGP << 320KB).
//   * Dual-accumulator GEMM phases (unconditional, branch-free) with bounded
//     unroll to avoid spills: one A-fragment feeds two WMMA tiles.
// ---------------------------------------------------------------------------

typedef _Float16 h16;
typedef __attribute__((ext_vector_type(16))) _Float16 v16h;
typedef __attribute__((ext_vector_type(8)))  _Float16 v8h;
typedef __attribute__((ext_vector_type(8)))  float    v8f;

#define B_TOT   64
#define MB      16              // rows per block (one WMMA M tile)
#define STOCH   128
#define DETER   512
#define EMBED   512
#define OBSE    512
#define NWAVE   16              // 512 threads = 16 wave32

// per-level transposed-f16 weight layout (element offsets)
#define O_PH1   0u              // [512 ,  640]
#define O_PH2   327680u         // [512 ,  512]
#define O_WI    589824u         // [1536,  512]
#define O_WH    1376256u        // [1536,  512]
#define O_QH1   2162688u        // [512 , 1024]
#define O_QH2   2686976u        // [512 ,  512]
#define O_QM    2949120u        // [128 ,  512]
#define O_QS    3014656u        // [128 ,  512]
#define WL      3080192u        // halfs per level
#define WT_BYTES (3u * WL * 2u) // 18,481,152 B

__device__ __forceinline__ v8f wmma16(v16h a, v16h b, v8f c) {
    return __builtin_amdgcn_wmma_f32_16x16x32_f16(
        /*neg_a*/false, a, /*neg_b*/false, b,
        /*c_mod*/(short)0, c, /*reuse_a*/false, /*reuse_b*/false);
}

// A fragment (16x32 f16) from row-major LDS tile, rows 0..15, cols k0..k0+31.
// ISA layout: lanes 0-15 hold K {0..7,16..23}, lanes 16-31 hold K {8..15,24..31}.
__device__ __forceinline__ v16h ldA(const h16* base, int stride, int k0, int lane) {
    int r  = lane & 15;
    int kk = (lane & 16) ? 8 : 0;
    const h16* p = base + r * stride + k0 + kk;
    v8h lo = *(const v8h*)p;
    v8h hi = *(const v8h*)(p + 16);
    v16h o;
#pragma unroll
    for (int i = 0; i < 8; ++i) { o[i] = lo[i]; o[i + 8] = hi[i]; }
    return o;
}

// B fragment (32x16 f16) from transposed weights Wt[N][K] in global memory.
// ISA layout: lanes 0-15 hold K {0..7,16..23} of col n, lanes 16-31 hold K {8..15,24..31}.
__device__ __forceinline__ v16h ldB(const h16* __restrict__ wt, int K,
                                    int n0, int k0, int lane) {
    int r  = lane & 15;
    int kk = (lane & 16) ? 8 : 0;
    const h16* p = wt + (size_t)(n0 + r) * K + k0 + kk;
    v8h lo = *(const v8h*)p;
    v8h hi = *(const v8h*)(p + 16);
    v16h o;
#pragma unroll
    for (int i = 0; i < 8; ++i) { o[i] = lo[i]; o[i + 8] = hi[i]; }
    return o;
}

// store 16x16 f32 accumulator tile (C/D layout) to row-major f16 LDS, +bias, opt relu
__device__ __forceinline__ void epiStore(h16* D, int stride, int n0, int lane,
                                         v8f acc, const float* __restrict__ bias,
                                         bool relu) {
    int n  = n0 + (lane & 15);
    int m0 = (lane & 16) ? 8 : 0;
    float b = bias[n];
#pragma unroll
    for (int v = 0; v < 8; ++v) {
        float x = acc[v] + b;
        if (relu) x = fmaxf(x, 0.0f);
        D[(m0 + v) * stride + n] = (h16)x;
    }
}

// M=16, N=512 GEMM phase:  D[16,512] = relu(A[16,K] @ Wt^T + bias)
// Exactly 32 N-tiles -> each wave owns tiles (wave, wave+16) unconditionally.
// One A fragment feeds both WMMA chains. Bounded unroll avoids spills.
__device__ __forceinline__ void mmPhase512(const h16* A, int As,
                                           const h16* __restrict__ W, int K,
                                           const float* __restrict__ bias,
                                           h16* D, bool relu,
                                           int wave, int lane) {
    const int n0 = wave << 4;
    const int n1 = (wave + NWAVE) << 4;
    v8f acc0 = {}, acc1 = {};
#pragma unroll 4
    for (int k0 = 0; k0 < K; k0 += 32) {
        v16h a = ldA(A, As, k0, lane);
        acc0 = wmma16(a, ldB(W, K, n0, k0, lane), acc0);
        acc1 = wmma16(a, ldB(W, K, n1, k0, lane), acc1);
    }
    epiStore(D, EMBED, n0, lane, acc0, bias, relu);
    epiStore(D, EMBED, n1, lane, acc1, bias, relu);
}

__device__ __forceinline__ float sigm(float x) { return 1.0f / (1.0f + __expf(-x)); }
__device__ __forceinline__ float softp(float x) {
    return (x > 20.0f) ? x : log1pf(__expf(x));
}

// f32 [K,N] -> f16 transposed [N,K]
__global__ void k_tconv(const float* __restrict__ src, h16* __restrict__ dst,
                        int K, int N) {
    int i = blockIdx.x * blockDim.x + threadIdx.x;
    if (i < K * N) {
        int k = i / N, n = i - k * N;
        dst[(size_t)n * K + k] = (h16)src[i];
    }
}

// ---------------------------------------------------------------------------
// Persistent per-level RSSM scan. grid = 4 blocks x 512 threads.
// Block b owns batch rows [16b, 16b+16). LDS buffers:
//   X  [16,640] : [prev_sample(128) | context(512)]  (later: q_sample write)
//   H1 [16,512] : h1 / h(temp) / hq1
//   H2 [16,512] : h2 / obs / hq2
//   HS [16,512] : prev_h (f16 state)
// ---------------------------------------------------------------------------
__global__ __launch_bounds__(512, 1)
void k_level(const h16* __restrict__ wt,
             const float* __restrict__ b_ph1, const float* __restrict__ b_ph2,
             const float* __restrict__ b_gi,  const float* __restrict__ b_gh,
             const float* __restrict__ b_qh1, const float* __restrict__ b_qh2,
             const float* __restrict__ b_qm,  const float* __restrict__ b_qs,
             const float* __restrict__ obs,   const float* __restrict__ nq,
             const float* __restrict__ ctx,   float* __restrict__ out, int T) {
    extern __shared__ h16 smem[];
    h16* X  = smem;                 // 16*640
    h16* H1 = X  + MB * 640;        // 16*512
    h16* H2 = H1 + MB * DETER;      // 16*512
    h16* HS = H2 + MB * DETER;      // 16*512

    const int tid  = threadIdx.x;
    const int wave = tid >> 5;
    const int lane = tid & 31;
    const int b0   = blockIdx.x * MB;
    const int Tc   = T >> 2;        // upper-level context length (FACTOR=4)

    const h16* Wph1 = wt + O_PH1;
    const h16* Wph2 = wt + O_PH2;
    const h16* Wwi  = wt + O_WI;
    const h16* Wwh  = wt + O_WH;
    const h16* Wqh1 = wt + O_QH1;
    const h16* Wqh2 = wt + O_QH2;
    const h16* Wqm  = wt + O_QM;
    const h16* Wqs  = wt + O_QS;

    // init carry: prev_sample = 0 (X cols 0..127), prev_h = 0
    for (int i = tid; i < MB * STOCH; i += 512) {
        int m = i >> 7, n = i & 127;
        X[m * 640 + n] = (h16)0.0f;
    }
    for (int i = tid; i < MB * DETER; i += 512) HS[i] = (h16)0.0f;
    __syncthreads();

    for (int t = 0; t < T; ++t) {
        // ---- P0: context(t) -> X[:,128:640] -------------------------------
        for (int i = tid; i < MB * DETER; i += 512) {
            int m = i >> 9, n = i & 511;
            float c = ctx ? ctx[((size_t)(b0 + m) * Tc + (t >> 2)) * DETER + n]
                          : 0.0f;
            X[m * 640 + 128 + n] = (h16)c;
        }
        __syncthreads();

        // ---- P1: h1 = relu([s|c] @ ph1_w + b) -----------------------------
        mmPhase512(X, 640, Wph1, 640, b_ph1, H1, true, wave, lane);
        __syncthreads();

        // ---- P2: h2 = relu(h1 @ ph2_w + b) --------------------------------
        mmPhase512(H1, EMBED, Wph2, EMBED, b_ph2, H2, true, wave, lane);
        __syncthreads();

        // ---- P3: GRU  (gi = h2 @ Wi, gh = prev_h @ Wh, fuse elementwise) --
        // 6 accumulation chains per tile share the two A fragments; kept
        // single-tile to stay under VGPR budget (no spills).
        for (int nt = wave; nt < (DETER >> 4); nt += NWAVE) {
            v8f ar = {}, az = {}, ag = {}, hr = {}, hz = {}, hg = {};
            for (int k0 = 0; k0 < EMBED; k0 += 32) {
                v16h a = ldA(H2, DETER, k0, lane);
                v16h p = ldA(HS, DETER, k0, lane);
                ar = wmma16(a, ldB(Wwi, EMBED, (nt << 4),        k0, lane), ar);
                az = wmma16(a, ldB(Wwi, EMBED, (nt << 4) + 512,  k0, lane), az);
                ag = wmma16(a, ldB(Wwi, EMBED, (nt << 4) + 1024, k0, lane), ag);
                hr = wmma16(p, ldB(Wwh, DETER, (nt << 4),        k0, lane), hr);
                hz = wmma16(p, ldB(Wwh, DETER, (nt << 4) + 512,  k0, lane), hz);
                hg = wmma16(p, ldB(Wwh, DETER, (nt << 4) + 1024, k0, lane), hg);
            }
            int n  = (nt << 4) + (lane & 15);
            int m0 = (lane & 16) ? 8 : 0;
            float bir = b_gi[n], biz = b_gi[512 + n], big = b_gi[1024 + n];
            float bhr = b_gh[n], bhz = b_gh[512 + n], bhg = b_gh[1024 + n];
#pragma unroll
            for (int v = 0; v < 8; ++v) {
                float r  = sigm(ar[v] + bir + hr[v] + bhr);
                float z  = sigm(az[v] + biz + hz[v] + bhz);
                float g  = tanhf(ag[v] + big + r * (hg[v] + bhg));
                float ph = (float)HS[(m0 + v) * DETER + n];
                float h  = (1.0f - z) * g + z * ph;
                H1[(m0 + v) * DETER + n] = (h16)h;
                out[((size_t)(b0 + m0 + v) * T + t) * DETER + n] = h;
            }
        }
        __syncthreads();

        // ---- P3b: HS <- h ; obs(t) -> H2 ----------------------------------
        for (int i = tid; i < MB * DETER; i += 512) {
            HS[i] = H1[i];
            int m = i >> 9, n = i & 511;
            H2[i] = (h16)obs[((size_t)(b0 + m) * T + t) * OBSE + n];
        }
        __syncthreads();

        // ---- P4: hq1 = relu([h|o] @ qh1_w + b) -> H1 (dual-tile, branch-free)
        {
            const int n0 = wave << 4;
            const int n1 = (wave + NWAVE) << 4;
            v8f acc0 = {}, acc1 = {};
#pragma unroll 4
            for (int k0 = 0; k0 < 512; k0 += 32) {
                v16h a = ldA(HS, DETER, k0, lane);
                acc0 = wmma16(a, ldB(Wqh1, 1024, n0, k0, lane), acc0);
                acc1 = wmma16(a, ldB(Wqh1, 1024, n1, k0, lane), acc1);
            }
#pragma unroll 4
            for (int k0 = 0; k0 < 512; k0 += 32) {
                v16h a = ldA(H2, DETER, k0, lane);
                acc0 = wmma16(a, ldB(Wqh1, 1024, n0, 512 + k0, lane), acc0);
                acc1 = wmma16(a, ldB(Wqh1, 1024, n1, 512 + k0, lane), acc1);
            }
            epiStore(H1, EMBED, n0, lane, acc0, b_qh1, true);
            epiStore(H1, EMBED, n1, lane, acc1, b_qh1, true);
        }
        __syncthreads();

        // ---- P5: hq2 = relu(hq1 @ qh2_w + b) -> H2 ------------------------
        mmPhase512(H1, EMBED, Wqh2, EMBED, b_qh2, H2, true, wave, lane);
        __syncthreads();

        // ---- P6: q_sample = (hq2@qm+b) + (softplus((hq2+.54)@qs+b)+1e-4)*nq
        if (wave < (STOCH >> 4)) {
            int nt = wave;
            v8f am = {}, as = {};
            for (int k0 = 0; k0 < EMBED; k0 += 32) {
                v16h a  = ldA(H2, DETER, k0, lane);
                v16h a2 = a + (_Float16)0.54f;
                am = wmma16(a,  ldB(Wqm, EMBED, nt << 4, k0, lane), am);
                as = wmma16(a2, ldB(Wqs, EMBED, nt << 4, k0, lane), as);
            }
            int n  = (nt << 4) + (lane & 15);
            int m0 = (lane & 16) ? 8 : 0;
            float bm = b_qm[n], bs = b_qs[n];
#pragma unroll
            for (int v = 0; v < 8; ++v) {
                float qm_ = am[v] + bm;
                float qs_ = softp(as[v] + bs) + 1e-4f;
                float nz  = nq[((size_t)(b0 + m0 + v) * T + t) * STOCH + n];
                X[(m0 + v) * 640 + n] = (h16)(qm_ + qs_ * nz);  // next prev_sample
            }
        }
        __syncthreads();
    }
}

// ---------------------------------------------------------------------------
// Host side.
// d_in order (JAX pytree sorted-key flatten of params, then obs/noise triples):
//  0 gru_bh  1 gru_bi  2 gru_wh  3 gru_wi  4 ph1_b  5 ph1_w  6 ph2_b  7 ph2_w
//  8 pm_b    9 pm_w   10 ps_b   11 ps_w   12 qh1_b 13 qh1_w 14 qh2_b 15 qh2_w
// 16 qm_b   17 qm_w   18 qs_b   19 qs_w
// 20 obs0 21 np0 22 nq0 | 23 obs1 24 np1 25 nq1 | 26 obs2 27 np2 28 nq2
// ---------------------------------------------------------------------------
extern "C" void kernel_launch(void* const* d_in, const int* in_sizes, int n_in,
                              void* d_out, int out_size, void* d_ws, size_t ws_size,
                              hipStream_t stream) {
    (void)in_sizes; (void)n_in; (void)out_size; (void)ws_size;

    const float* gru_bh = (const float*)d_in[0];
    const float* gru_bi = (const float*)d_in[1];
    const float* gru_wh = (const float*)d_in[2];
    const float* gru_wi = (const float*)d_in[3];
    const float* ph1_b  = (const float*)d_in[4];
    const float* ph1_w  = (const float*)d_in[5];
    const float* ph2_b  = (const float*)d_in[6];
    const float* ph2_w  = (const float*)d_in[7];
    const float* qh1_b  = (const float*)d_in[12];
    const float* qh1_w  = (const float*)d_in[13];
    const float* qh2_b  = (const float*)d_in[14];
    const float* qh2_w  = (const float*)d_in[15];
    const float* qm_b   = (const float*)d_in[16];
    const float* qm_w   = (const float*)d_in[17];
    const float* qs_b   = (const float*)d_in[18];
    const float* qs_w   = (const float*)d_in[19];
    const float* obsL[3] = { (const float*)d_in[20], (const float*)d_in[23],
                             (const float*)d_in[26] };
    const float* nqL[3]  = { (const float*)d_in[22], (const float*)d_in[25],
                             (const float*)d_in[28] };
    const int TL[3] = { 256, 64, 16 };

    h16* wt = (h16*)d_ws;
    auto tp = [&](const float* s, h16* d, int K, int N) {
        int n = K * N;
        k_tconv<<<(n + 255) / 256, 256, 0, stream>>>(s, d, K, N);
    };
    for (int L = 0; L < 3; ++L) {
        h16* w = wt + (size_t)L * WL;
        tp(ph1_w  + (size_t)L * 640  * 512,  w + O_PH1, 640,  512);
        tp(ph2_w  + (size_t)L * 512  * 512,  w + O_PH2, 512,  512);
        tp(gru_wi + (size_t)L * 512  * 1536, w + O_WI,  512,  1536);
        tp(gru_wh + (size_t)L * 512  * 1536, w + O_WH,  512,  1536);
        tp(qh1_w  + (size_t)L * 1024 * 512,  w + O_QH1, 1024, 512);
        tp(qh2_w  + (size_t)L * 512  * 512,  w + O_QH2, 512,  512);
        tp(qm_w   + (size_t)L * 512  * 128,  w + O_QM,  512,  128);
        tp(qs_w   + (size_t)L * 512  * 128,  w + O_QS,  512,  128);
    }

    // context buffers: level-2 output (B x 16 x 512), level-1 output (B x 64 x 512)
    float* ctx1 = (float*)((char*)d_ws + WT_BYTES);
    float* ctx0 = ctx1 + (size_t)B_TOT * 16 * DETER;

    const size_t smemBytes = (size_t)(MB * 640 + 3 * MB * DETER) * sizeof(h16);

    auto runLevel = [&](int L, const float* ctx, float* out) {
        const h16* w = wt + (size_t)L * WL;
        k_level<<<B_TOT / MB, NWAVE * 32, smemBytes, stream>>>(
            w,
            ph1_b + (size_t)L * 512, ph2_b + (size_t)L * 512,
            gru_bi + (size_t)L * 1536, gru_bh + (size_t)L * 1536,
            qh1_b + (size_t)L * 512, qh2_b + (size_t)L * 512,
            qm_b + (size_t)L * 128, qs_b + (size_t)L * 128,
            obsL[L], nqL[L], ctx, out, TL[L]);
    };

    runLevel(2, nullptr, ctx1);          // top: zero context
    runLevel(1, ctx1, ctx0);             // ctx repeated x4 via t>>2 indexing
    runLevel(0, ctx0, (float*)d_out);    // final h sequence -> d_out
}
